// GraphEncoder_63634235457844
// MI455X (gfx1250) — matrix operand order repaired and checked by reading, not compiled, not run on hardware
//
#include <hip/hip_runtime.h>
#include <hip/hip_bf16.h>

// ---------------- problem constants ----------------
#define NN       100000     // nodes
#define NE       1600000    // edges
#define NG       256        // graphs
#define INC      64
#define CC       128
#define NLAYERS  5
#define LN_EPS   1e-5f

typedef __attribute__((ext_vector_type(16))) __bf16   v16bf;
typedef __attribute__((ext_vector_type(8)))  float    v8f;
typedef __attribute__((ext_vector_type(4)))  unsigned v4u;
typedef __attribute__((ext_vector_type(8)))  int      v8i;
typedef __attribute__((ext_vector_type(4)))  int      v4i;

// f32 -> bf16 round-to-nearest-even (bit manipulation; avoids relying on __bf16 conversions)
__device__ __forceinline__ __bf16 f2bf(float f) {
    union { float f; unsigned u; } cv; cv.f = f;
    unsigned r = cv.u + 0x7FFFu + ((cv.u >> 16) & 1u);
    union { unsigned short s; __bf16 b; } o; o.s = (unsigned short)(r >> 16);
    return o.b;
}

// ---- WMMA fragment loaders (CDNA5 16x16x32 bf16 layouts, ISA 7.12.2) ----
// A: 16xK tile in LDS, row-major, K contiguous (stride = K elems). Lane L: M=L&15, half=L>>4.
__device__ __forceinline__ v16bf load_a_frag(const __bf16* tile, int stride, int kchunk, int lane) {
    const int M = lane & 15, half = lane >> 4;
    const unsigned* t32 = (const unsigned*)(tile + M * stride + kchunk * 32);
    v16bf a;
#pragma unroll
    for (int v = 0; v < 8; ++v) {
        const int kb = (v < 4) ? (half * 8 + v * 2) : (16 + half * 8 + (v - 4) * 2);
        union { unsigned u; __bf16 h[2]; } cv; cv.u = t32[kb >> 1];
        a[2 * v]     = cv.h[0];
        a[2 * v + 1] = cv.h[1];
    }
    return a;
}
// B: weights stored transposed in LDS: wT[n*Kstride + k] (k contiguous per output channel n).
__device__ __forceinline__ v16bf load_b_frag(const __bf16* wT, int Kstride, int kchunk, int lane) {
    const int N = lane & 15, half = lane >> 4;
    const unsigned* t32 = (const unsigned*)(wT + N * Kstride + kchunk * 32);
    v16bf b;
#pragma unroll
    for (int v = 0; v < 8; ++v) {
        const int kb = (v < 4) ? (half * 8 + v * 2) : (16 + half * 8 + (v - 4) * 2);
        union { unsigned u; __bf16 h[2]; } cv; cv.u = t32[kb >> 1];
        b[2 * v]     = cv.h[0];
        b[2 * v + 1] = cv.h[1];
    }
    return b;
}

// ---- TDM: async 2-D bf16 tile load global->LDS (D# per ISA cdna5 §8.3/8.4) ----
// dim0 = elements per row (contiguous), dim1 = rows; tile == tensor (full copy).
// amdgpu-toolchain (clang-23) builtin form: 6 args (g0, g1, g2, g3, g?, cpol).
__device__ __forceinline__ void tdm_load_2d_bf16(const __bf16* gsrc, unsigned lds_byte_addr,
                                                 int dim0, int dim1) {
    const unsigned long long ga = (unsigned long long)(uintptr_t)gsrc;
    v4u g0;
    g0[0] = 1u;                                                  // count=1, user descriptor
    g0[1] = lds_byte_addr;                                       // lds_addr [63:32]
    g0[2] = (unsigned)(ga & 0xFFFFFFFFu);                        // global_addr[31:0]
    g0[3] = (unsigned)((ga >> 32) & 0x01FFFFFFu) | (2u << 30);   // addr[56:32] | type=2
    v8i g1;
    g1[0] = (int)(1u << 16);                                     // data_size=1 (2 bytes)
    g1[1] = (int)((unsigned)(dim0 & 0xFFFF) << 16);              // tensor_dim0 lo16 @ bits 63:48
    g1[2] = (int)((unsigned)(dim1 & 0xFFFF) << 16);              // tensor_dim1 lo16 @ bits 111:80
    g1[3] = (int)((unsigned)(dim0 & 0xFFFF) << 16);              // tile_dim0 @ bits 127:112
    g1[4] = (int)((unsigned)(dim1 & 0xFFFF));                    // tile_dim1 @ bits 143:128
    g1[5] = dim0;                                                // tensor_dim0_stride lo32
    g1[6] = 0;
    g1[7] = 0;
    const v4i z4 = {0, 0, 0, 0};                                 // groups 2/3 disabled (2-D)
    const v8i z8 = {0, 0, 0, 0, 0, 0, 0, 0};
    __builtin_amdgcn_tensor_load_to_lds(g0, g1, z4, z4, z8, 0);
}

// ---------------- weight prep: f32 -> bf16, transposed to [n][k] ----------------
__global__ void prep_weights(const float* __restrict__ proj_w,   // [64][128]
                             const float* __restrict__ gw1,      // [5][128][128]
                             const float* __restrict__ gw2,      // [5][128][128]
                             __bf16* __restrict__ projT,         // [128][64]
                             __bf16* __restrict__ w1T,           // [5][128][128] (n-major)
                             __bf16* __restrict__ w2T) {
    const int stride = gridDim.x * blockDim.x;
    int idx = blockIdx.x * blockDim.x + threadIdx.x;
    for (int i = idx; i < CC * INC; i += stride) {
        int n = i >> 6, k = i & 63;
        projT[i] = f2bf(proj_w[k * CC + n]);
    }
    for (int i = idx; i < NLAYERS * CC * CC; i += stride) {
        int l = i >> 14, r = i & 16383;
        int n = r >> 7, k = r & 127;
        w1T[i] = f2bf(gw1[(l << 14) + (k << 7) + n]);
        w2T[i] = f2bf(gw2[(l << 14) + (k << 7) + n]);
    }
}

// ---------------- projection: h = x @ proj_w + proj_b ----------------
__global__ void proj_kernel(const float* __restrict__ x, const __bf16* __restrict__ projT,
                            const float* __restrict__ projb, float* __restrict__ h) {
    __shared__ __bf16 pw[CC * INC];   // 16 KB
    __shared__ __bf16 xt[16 * INC];   //  2 KB
    const int t = threadIdx.x, lane = t & 31, wv = t >> 5;
    for (int i = t; i < CC * INC; i += 256) pw[i] = projT[i];
    for (int sub = 0; sub < 8; ++sub) {
        const int base = blockIdx.x * 128 + sub * 16;
        if (base >= NN) break;                       // NN % 16 == 0, uniform
        __syncthreads();
        for (int i = t; i < 16 * INC; i += 256) {
            int r = i >> 6, c = i & 63;
            xt[i] = f2bf(x[(size_t)(base + r) * INC + c]);
        }
        __syncthreads();
        v8f acc = {};
#pragma unroll
        for (int kc = 0; kc < 2; ++kc) {
            v16bf a = load_a_frag(xt, INC, kc, lane);
            v16bf b = load_b_frag(pw + wv * 16 * INC, INC, kc, lane);
            acc = __builtin_amdgcn_wmma_f32_16x16x32_bf16(false, a, false, b, (short)0, acc, false, false);
        }
        const int n = wv * 16 + (lane & 15), half = lane >> 4;
        const float bia = projb[n];
#pragma unroll
        for (int r = 0; r < 8; ++r) {
            const int m = r + 8 * half;
            h[(size_t)(base + m) * CC + n] = acc[r] + bia;
        }
    }
}

// ---------------- edge aggregation: z[dst] += h[src] (z preloaded with h) ----------------
// Block stages 64 edges' indices in LDS (kills the 32x redundant index fetch),
// then 256 threads process 64 edges x 32 float4-chunks.
__global__ void edge_aggr(const int* __restrict__ src, const int* __restrict__ dst,
                          const float* __restrict__ h, float* __restrict__ z) {
    __shared__ int se[64], de[64];
    const long long ebase = (long long)blockIdx.x * 64;
    const int t = threadIdx.x;
    if (t < 64) {
        const long long e = ebase + t;
        se[t] = (e < NE) ? src[e] : -1;
        de[t] = (e < NE) ? dst[e] : 0;
    }
    __syncthreads();
#pragma unroll 2
    for (int i = t; i < 64 * 32; i += 256) {
        const int le = i >> 5;
        const int s = se[le];
        if (s < 0) continue;
        const int c4 = (i & 31) * 4;
        const float4 v = *(const float4*)(h + (size_t)s * CC + c4);
        float* zp = z + (size_t)de[le] * CC + c4;
        __builtin_prefetch(zp, 1, 1);                 // global_prefetch
        atomicAdd(zp + 0, v.x);
        atomicAdd(zp + 1, v.y);
        atomicAdd(zp + 2, v.z);
        atomicAdd(zp + 3, v.w);
    }
}

// ---------------- fused GIN MLP + LayerNorm ----------------
// block = 256 thr (8 waves), 128 nodes per block in 8 subtiles of 16.
// dynamic LDS partition (84096 B total); weights staged via TDM.
#define SM_W1   0
#define SM_W2   32768
#define SM_ZT   65536
#define SM_MID  69632
#define SM_Z2   73728
#define SM_RED  81920
#define SM_MRS  83968
#define SMEM_MLP 84096

__global__ void mlp_ln_kernel(const __bf16* __restrict__ w1T, const float* __restrict__ b1g,
                              const __bf16* __restrict__ w2T, const float* __restrict__ b2g,
                              const float* __restrict__ lng,  const float* __restrict__ lnb,
                              const float* __restrict__ zsrc, float* __restrict__ hdst) {
    extern __shared__ char smem[];
    __bf16* w1s = (__bf16*)(smem + SM_W1);
    __bf16* w2s = (__bf16*)(smem + SM_W2);
    __bf16* zt  = (__bf16*)(smem + SM_ZT);
    __bf16* mid = (__bf16*)(smem + SM_MID);
    float*  z2  = (float*) (smem + SM_Z2);
    float*  red = (float*) (smem + SM_RED);   // 512 floats
    float*  mrs = (float*) (smem + SM_MRS);   // mu[16], rstd[16]

    const int t = threadIdx.x, lane = t & 31, wv = t >> 5;

    // Async weight staging via Tensor Data Mover (wave 0 issues; TENSORcnt is per-wave)
    if (wv == 0) {
        tdm_load_2d_bf16(w1T, (unsigned)(uintptr_t)w1s, CC, CC);
        tdm_load_2d_bf16(w2T, (unsigned)(uintptr_t)w2s, CC, CC);
        __builtin_amdgcn_s_wait_tensorcnt(0);
    }

    const int n = wv * 16 + (lane & 15), half = lane >> 4;
    const float bb1 = b1g[n], bb2 = b2g[n];

    for (int sub = 0; sub < 8; ++sub) {
        const int base = blockIdx.x * 128 + sub * 16;
        if (base >= NN) break;
        __syncthreads();   // first iter: publishes TDM-loaded weights to all waves
        for (int i = t; i < 16 * CC; i += 256) {
            int r = i >> 7, c = i & 127;
            zt[i] = f2bf(zsrc[(size_t)(base + r) * CC + c]);
        }
        __syncthreads();

        // GEMM1 + bias + relu -> mid (bf16 LDS)
        v8f acc = {};
#pragma unroll
        for (int kc = 0; kc < 4; ++kc) {
            v16bf a = load_a_frag(zt, CC, kc, lane);
            v16bf b = load_b_frag(w1s + wv * 16 * CC, CC, kc, lane);
            acc = __builtin_amdgcn_wmma_f32_16x16x32_bf16(false, a, false, b, (short)0, acc, false, false);
        }
#pragma unroll
        for (int r = 0; r < 8; ++r) {
            const int m = r + 8 * half;
            mid[m * CC + n] = f2bf(fmaxf(acc[r] + bb1, 0.f));
        }
        __syncthreads();

        // GEMM2 + bias + relu -> z2 (f32 LDS)
        v8f acc2 = {};
#pragma unroll
        for (int kc = 0; kc < 4; ++kc) {
            v16bf a = load_a_frag(mid, CC, kc, lane);
            v16bf b = load_b_frag(w2s + wv * 16 * CC, CC, kc, lane);
            acc2 = __builtin_amdgcn_wmma_f32_16x16x32_bf16(false, a, false, b, (short)0, acc2, false, false);
        }
#pragma unroll
        for (int r = 0; r < 8; ++r) {
            const int m = r + 8 * half;
            z2[m * CC + n] = fmaxf(acc2[r] + bb2, 0.f);
        }
        __syncthreads();

        // LayerNorm over 128 channels per row
        {
            const int row = t >> 4, sg = t & 15;
            float s = 0.f, ss = 0.f;
#pragma unroll
            for (int j = 0; j < 8; ++j) {
                float v = z2[row * CC + sg * 8 + j];
                s += v; ss += v * v;
            }
            red[t] = s; red[256 + t] = ss;
        }
        __syncthreads();
        if (t < 16) {
            float s = 0.f, ss = 0.f;
            for (int j = 0; j < 16; ++j) { s += red[t * 16 + j]; ss += red[256 + t * 16 + j]; }
            const float mu  = s * (1.f / 128.f);
            const float var = ss * (1.f / 128.f) - mu * mu;
            mrs[t] = mu; mrs[16 + t] = rsqrtf(var + LN_EPS);
        }
        __syncthreads();
        for (int i = t; i < 16 * CC; i += 256) {
            int r = i >> 7, c = i & 127;
            float v = (z2[i] - mrs[r]) * mrs[16 + r] * lng[c] + lnb[c];
            hdst[(size_t)(base + r) * CC + c] = v;
        }
    }
}

// ---------------- pooling ----------------
__global__ void zero_g(float* __restrict__ g) {
    int i = blockIdx.x * blockDim.x + threadIdx.x;
    if (i < NG * CC) g[i] = 0.f;
}
// Block stages 64 nodes' batch ids in LDS, then 64 nodes x 32 chunks over 256 threads.
__global__ void pool_kernel(const int* __restrict__ batch, const float* __restrict__ h,
                            float* __restrict__ g) {
    __shared__ int bn[64];
    const long long nbase = (long long)blockIdx.x * 64;
    const int t = threadIdx.x;
    if (t < 64) {
        const long long nid = nbase + t;
        bn[t] = (nid < NN) ? batch[nid] : -1;
    }
    __syncthreads();
#pragma unroll 2
    for (int i = t; i < 64 * 32; i += 256) {
        const int ln = i >> 5;
        const int b = bn[ln];
        if (b < 0) continue;
        const int c4 = (i & 31) * 4;
        const float4 v = *(const float4*)(h + (size_t)(nbase + ln) * CC + c4);
        float* gp = g + (size_t)b * CC + c4;
        atomicAdd(gp + 0, v.x);
        atomicAdd(gp + 1, v.y);
        atomicAdd(gp + 2, v.z);
        atomicAdd(gp + 3, v.w);
    }
}

// ---------------- final MLP (tiny: 33 MFLOP total, scalar f32) ----------------
__global__ void final_mlp(const float* __restrict__ g,
                          const float* __restrict__ fw1, const float* __restrict__ fb1,
                          const float* __restrict__ fw2, const float* __restrict__ fb2,
                          float* __restrict__ out) {
    __shared__ float gr[CC];
    __shared__ float hid[2 * CC];
    const int gid = blockIdx.x, t = threadIdx.x;
    if (t < CC) gr[t] = g[(size_t)gid * CC + t];
    __syncthreads();
    float s = fb1[t];
    for (int k = 0; k < CC; ++k) s += gr[k] * fw1[k * 2 * CC + t];
    hid[t] = fmaxf(s, 0.f);
    __syncthreads();
    if (t < CC) {
        float o = fb2[t];
        for (int k = 0; k < 2 * CC; ++k) o += hid[k] * fw2[k * CC + t];
        out[(size_t)gid * CC + t] = o;
    }
}

// ---------------- launcher ----------------
extern "C" void kernel_launch(void* const* d_in, const int* in_sizes, int n_in,
                              void* d_out, int out_size, void* d_ws, size_t ws_size,
                              hipStream_t stream) {
    const float* x      = (const float*)d_in[0];
    const int*   ei     = (const int*)  d_in[1];
    const int*   batch  = (const int*)  d_in[2];
    const float* proj_w = (const float*)d_in[3];
    const float* proj_b = (const float*)d_in[4];
    const float* gw1    = (const float*)d_in[5];
    const float* gb1    = (const float*)d_in[6];
    const float* gw2    = (const float*)d_in[7];
    const float* gb2    = (const float*)d_in[8];
    const float* lng    = (const float*)d_in[9];
    const float* lnb    = (const float*)d_in[10];
    const float* fw1    = (const float*)d_in[11];
    const float* fb1    = (const float*)d_in[12];
    const float* fw2    = (const float*)d_in[13];
    const float* fb2    = (const float*)d_in[14];
    float* out = (float*)d_out;

    const int* src = ei;
    const int* dst = ei + NE;

    // workspace partition
    char* ws = (char*)d_ws;
    const size_t HB = (size_t)NN * CC * sizeof(float);        // 51.2 MB
    float*  h     = (float*)(ws);
    float*  z     = (float*)(ws + HB);
    __bf16* projT = (__bf16*)(ws + 2 * HB);
    __bf16* w1T   = (__bf16*)(ws + 2 * HB + 16640);
    __bf16* w2T   = (__bf16*)(ws + 2 * HB + 16640 + 163840);
    float*  g     = (float*) (ws + 2 * HB + 16640 + 2 * 163840);

    const int NODE_BLOCKS = (NN + 127) / 128;                  // 782

    prep_weights<<<160, 256, 0, stream>>>(proj_w, gw1, gw2, projT, w1T, w2T);
    proj_kernel<<<NODE_BLOCKS, 256, 0, stream>>>(x, projT, proj_b, h);

    const int edge_blocks = (NE + 63) / 64;                    // 25000

    for (int l = 0; l < NLAYERS; ++l) {
        (void)hipMemcpyAsync(z, h, HB, hipMemcpyDeviceToDevice, stream);   // z = h (self term)
        edge_aggr<<<edge_blocks, 256, 0, stream>>>(src, dst, h, z);
        mlp_ln_kernel<<<NODE_BLOCKS, 256, SMEM_MLP, stream>>>(
            w1T + (size_t)l * CC * CC, gb1 + l * CC,
            w2T + (size_t)l * CC * CC, gb2 + l * CC,
            lng + l * CC, lnb + l * CC, z, h);
    }

    zero_g<<<(NG * CC + 255) / 256, 256, 0, stream>>>(g);
    pool_kernel<<<(NN + 63) / 64, 256, 0, stream>>>(batch, h, g);
    final_mlp<<<NG, 256, 0, stream>>>(g, fw1, fb1, fw2, fb2, out);
    (void)in_sizes; (void)n_in; (void)out_size; (void)ws_size;
}
